// VectorQuantizer_352187319226
// MI455X (gfx1250) — matrix operand order repaired and checked by reading, not compiled, not run on hardware
//
#include <hip/hip_runtime.h>

// ---------------------------------------------------------------------------
// VQ codebook search on MI455X (gfx1250), wave32 + WMMA bf16 split-precision.
//   d_in[0] : z      [131072 x 256] f32
//   d_in[1] : weight [4096   x 256] f32
//   d_out   : z_q [131072 x 256] f32  ++  indices [131072] (stored as f32)
//   d_ws    : w_hi bf16[4096x256] | w_lo bf16[4096x256] | w_sq f32[4096]
//
// Main kernel: 8 wave32s/block, 2 row-strips (32 z-rows) per wave => 256
// rows/block (halves weight L2 re-read traffic vs 128). Weight tiles are
// double-buffered in LDS via CDNA5 async global->LDS copies (ASYNCcnt).
// ---------------------------------------------------------------------------

#define DIM        256
#define KCODES     4096
#define TILE_K     16                 // codes per WMMA column tile
#define NTILES     (KCODES / TILE_K)  // 256
#define BLOCK_THREADS 256             // 8 wave32s
#define STRIPS     2
#define ROWS_PER_WAVE (16 * STRIPS)   // 32
#define ROWS_PER_BLOCK (8 * ROWS_PER_WAVE) // 256

typedef __attribute__((ext_vector_type(16))) __bf16 v16bf;
typedef __attribute__((ext_vector_type(8)))  float  v8f;
typedef __attribute__((ext_vector_type(4)))  int    v4i;

__device__ __forceinline__ unsigned short f32_to_bf16_rne(float x) {
    unsigned int u = __float_as_uint(x);
    unsigned int r = u + 0x7FFFu + ((u >> 16) & 1u);
    return (unsigned short)(r >> 16);
}
__device__ __forceinline__ float bf16_to_f32(unsigned short h) {
    return __uint_as_float(((unsigned int)h) << 16);
}

__device__ __forceinline__ v8f wmma_bf16(v16bf a, v16bf b, v8f c) {
#if defined(__HIP_DEVICE_COMPILE__)
    // (neg_a, A, neg_b, B, c_mod, C, reuse_a, reuse_b)
    return __builtin_amdgcn_wmma_f32_16x16x32_bf16(
        false, a, false, b, (short)0, c, false, false);
#else
    return c;
#endif
}

// ---- CDNA5 async global->LDS copy (ASYNCcnt-tracked) ----------------------
#if defined(__HIP_DEVICE_COMPILE__)
# if __has_builtin(__builtin_amdgcn_global_load_async_to_lds_b128)
#   define VQ_ASYNC_BUILTIN 1
# else
#   define VQ_ASYNC_ASM 1
# endif
#endif

#if defined(VQ_ASYNC_BUILTIN)
typedef __attribute__((address_space(1))) v4i* gv4i_p;   // global int4*
typedef __attribute__((address_space(3))) v4i* lv4i_p;   // LDS int4*
#endif

__device__ __forceinline__ void async_copy_b128(const void* g, void* l) {
#if defined(VQ_ASYNC_BUILTIN)
    __builtin_amdgcn_global_load_async_to_lds_b128(
        (gv4i_p)g, (lv4i_p)l, 0, 0);
#elif defined(VQ_ASYNC_ASM)
    // VDST = VGPR holding LDS byte address, VADDR = 64-bit global address.
    unsigned loff = (unsigned)(unsigned long long)(uintptr_t)l;
    unsigned long long ga = (unsigned long long)(uintptr_t)g;
    asm volatile("global_load_async_to_lds_b128 %0, %1, off"
                 :: "v"(loff), "v"(ga) : "memory");
#else
    (void)g; (void)l;
#endif
}

__device__ __forceinline__ void wait_async0() {
#if defined(__HIP_DEVICE_COMPILE__)
# if __has_builtin(__builtin_amdgcn_s_wait_asynccnt)
    __builtin_amdgcn_s_wait_asynccnt(0);
# else
    asm volatile("s_wait_asynccnt 0x0" ::: "memory");
# endif
#endif
}

union FragU {
    v16bf v;
    unsigned short s[16];
};

// ---------------------------------------------------------------------------
// Kernel 1: weight f32 -> bf16 hi/lo split + per-code squared norm.
// ---------------------------------------------------------------------------
__global__ void vq_prep_weight(const float* __restrict__ w,
                               unsigned short* __restrict__ w_hi,
                               unsigned short* __restrict__ w_lo,
                               float* __restrict__ w_sq) {
    const int k = blockIdx.x;
    const int d = threadIdx.x;
    const int lane = threadIdx.x & 31;
    const int wid  = threadIdx.x >> 5;

    float x = w[k * DIM + d];
    unsigned short hb = f32_to_bf16_rne(x);
    float hi = bf16_to_f32(hb);
    unsigned short lb = f32_to_bf16_rne(x - hi);
    w_hi[k * DIM + d] = hb;
    w_lo[k * DIM + d] = lb;

    float s = x * x;
    #pragma unroll
    for (int m = 16; m > 0; m >>= 1) s += __shfl_xor(s, m, 32);

    __shared__ float red[8];
    if (lane == 0) red[wid] = s;
    __syncthreads();
    if (threadIdx.x == 0) {
        float t = 0.f;
        #pragma unroll
        for (int i = 0; i < 8; ++i) t += red[i];
        w_sq[k] = t;
    }
}

// ---------------------------------------------------------------------------
// Kernel 2: main search.
// ---------------------------------------------------------------------------
__global__ void __launch_bounds__(BLOCK_THREADS)
vq_search(const float* __restrict__ z,
          const float* __restrict__ weight,
          const unsigned short* __restrict__ w_hi,
          const unsigned short* __restrict__ w_lo,
          const float* __restrict__ w_sq,
          float* __restrict__ zq,
          float* __restrict__ out_idx,
          int n_rows) {
    const int tid  = threadIdx.x;
    const int lane = tid & 31;
    const int wid  = tid >> 5;
    const int half = lane >> 4;     // 0: lanes 0-15, 1: lanes 16-31
    const int l16  = lane & 15;

    const int rowBase = blockIdx.x * ROWS_PER_BLOCK + wid * ROWS_PER_WAVE;

    // ---- Register-resident A fragments: 2 strips x (16 rows x 256 d) ----
    // 16-bit A layout used here (16x32, contiguous k): lane half h, row M = lane%16,
    //   e0..e15 <-> K = kk + 16h + 0..15
    v16bf a_hi[STRIPS][8], a_lo[STRIPS][8];
    #pragma unroll
    for (int s = 0; s < STRIPS; ++s) {
        const float* zrow = z + (size_t)(rowBase + 16 * s + l16) * DIM;
        #pragma unroll
        for (int c = 0; c < 8; ++c) {
            const int kk = c * 32;
            float4 q0 = *(const float4*)(zrow + kk + 16 * half);
            float4 q1 = *(const float4*)(zrow + kk + 16 * half + 4);
            float4 q2 = *(const float4*)(zrow + kk + 16 * half + 8);
            float4 q3 = *(const float4*)(zrow + kk + 16 * half + 12);
            float xs[16] = {q0.x, q0.y, q0.z, q0.w, q1.x, q1.y, q1.z, q1.w,
                            q2.x, q2.y, q2.z, q2.w, q3.x, q3.y, q3.z, q3.w};
            FragU fh, fl;
            #pragma unroll
            for (int i = 0; i < 16; ++i) {
                unsigned short hb = f32_to_bf16_rne(xs[i]);
                fh.s[i] = hb;
                fl.s[i] = f32_to_bf16_rne(xs[i] - bf16_to_f32(hb));
            }
            a_hi[s][c] = fh.v;
            a_lo[s][c] = fl.v;
        }
    }

    // ---- LDS double buffer for weight tiles: [buf][hi/lo][16*256 bf16] ----
    __shared__ unsigned short sh[2][2][TILE_K * DIM];   // 32 KB

    // Async staging: 8KB hi + 8KB lo per tile; each thread issues 2+2 b128.
    auto stage_async = [&](int buf, int kt) {
        const unsigned short* gh = w_hi + (size_t)kt * TILE_K * DIM + tid * 8;
        const unsigned short* gl = w_lo + (size_t)kt * TILE_K * DIM + tid * 8;
        unsigned short* lh = &sh[buf][0][tid * 8];
        unsigned short* ll = &sh[buf][1][tid * 8];
        async_copy_b128(gh,        lh);
        async_copy_b128(gh + 2048, lh + 2048);
        async_copy_b128(gl,        ll);
        async_copy_b128(gl + 2048, ll + 2048);
    };

    float runmin[STRIPS][8];
    int   runidx[STRIPS][8];
    #pragma unroll
    for (int s = 0; s < STRIPS; ++s)
        #pragma unroll
        for (int r = 0; r < 8; ++r) { runmin[s][r] = 3.4e38f; runidx[s][r] = 0; }

    stage_async(0, 0);
    wait_async0();
    __syncthreads();

    for (int kt = 0; kt < NTILES; ++kt) {
        const int buf = kt & 1;
        if (kt + 1 < NTILES) stage_async(buf ^ 1, kt + 1);  // overlaps WMMA

        // B fragment (32x16 bf16): lane = column N = lane%16,
        // half selects K range: e0..e15 <-> K = kk + 16*half + 0..15.
        const unsigned short* th = &sh[buf][0][0];
        const unsigned short* tl = &sh[buf][1][0];
        const int boff = l16 * DIM + 16 * half;

        v8f acc0 = {}, acc1 = {};
        #pragma unroll
        for (int c = 0; c < 8; ++c) {
            const int kk = c * 32;
            v16bf bh = *(const v16bf*)(th + boff + kk);
            v16bf bl = *(const v16bf*)(tl + boff + kk);
            acc0 = wmma_bf16(a_hi[0][c], bh, acc0);   // hi*hi
            acc0 = wmma_bf16(a_hi[0][c], bl, acc0);   // hi*lo
            acc0 = wmma_bf16(a_lo[0][c], bh, acc0);   // lo*hi
            acc1 = wmma_bf16(a_hi[1][c], bh, acc1);
            acc1 = wmma_bf16(a_hi[1][c], bl, acc1);
            acc1 = wmma_bf16(a_lo[1][c], bh, acc1);
        }

        // score = ||w||^2 - 2 z.w ; lane's column code is fixed per tile.
        const int code = kt * TILE_K + l16;
        const float wsq = w_sq[code];
        #pragma unroll
        for (int r = 0; r < 8; ++r) {
            float s0 = fmaf(-2.f, acc0[r], wsq);
            if (s0 < runmin[0][r]) { runmin[0][r] = s0; runidx[0][r] = code; }
            float s1 = fmaf(-2.f, acc1[r], wsq);
            if (s1 < runmin[1][r]) { runmin[1][r] = s1; runidx[1][r] = code; }
        }

        wait_async0();      // our prefetch of buf^1 has landed in LDS
        __syncthreads();    // everyone done reading buf / prefetch visible
    }

    // ---- Cross-lane argmin within each 16-lane group (rows r / r+8) ----
    #pragma unroll
    for (int s = 0; s < STRIPS; ++s) {
        #pragma unroll
        for (int r = 0; r < 8; ++r) {
            float m = runmin[s][r];
            int   id = runidx[s][r];
            #pragma unroll
            for (int mask = 1; mask < 16; mask <<= 1) {
                float om = __shfl_xor(m, mask, 16);
                int   oid = __shfl_xor(id, mask, 16);
                if (om < m || (om == m && oid < id)) { m = om; id = oid; }
            }
            runmin[s][r] = m;
            runidx[s][r] = id;
        }
    }

    // ---- Gather z_q = weight[idx] (exact f32) and emit indices ----
    #pragma unroll
    for (int mrow = 0; mrow < ROWS_PER_WAVE; ++mrow) {
        const int s = mrow >> 4;
        const int r = mrow & 7;
        const int srcLane = (mrow & 8) ? 16 : 0;
        const int idx = __shfl(runidx[s][r], srcLane, 32);
        const int row = rowBase + mrow;
        if (row < n_rows) {
            const float4* wsrc = (const float4*)(weight + (size_t)idx * DIM);
            float4* dst = (float4*)(zq + (size_t)row * DIM);
            dst[lane * 2]     = wsrc[lane * 2];
            dst[lane * 2 + 1] = wsrc[lane * 2 + 1];
            if (lane == 0) out_idx[row] = (float)idx;
        }
    }
}

// ---------------------------------------------------------------------------
extern "C" void kernel_launch(void* const* d_in, const int* in_sizes, int n_in,
                              void* d_out, int out_size, void* d_ws, size_t ws_size,
                              hipStream_t stream) {
    const float* z      = (const float*)d_in[0];
    const float* weight = (const float*)d_in[1];
    const int n_rows = in_sizes[0] / DIM;      // 131072

    // workspace carve-up (~4.02 MB)
    unsigned short* w_hi = (unsigned short*)d_ws;
    unsigned short* w_lo = w_hi + (size_t)KCODES * DIM;
    float*          w_sq = (float*)(w_lo + (size_t)KCODES * DIM);

    float* zq      = (float*)d_out;
    float* out_idx = zq + (size_t)n_rows * DIM;

    vq_prep_weight<<<KCODES, 256, 0, stream>>>(weight, w_hi, w_lo, w_sq);

    const int grid = (n_rows + ROWS_PER_BLOCK - 1) / ROWS_PER_BLOCK;  // 512
    vq_search<<<grid, BLOCK_THREADS, 0, stream>>>(
        z, weight, w_hi, w_lo, w_sq, zq, out_idx, n_rows);
}
